// TransferBlock_26792005992640
// MI455X (gfx1250) — matrix operand order repaired and checked by reading, not compiled, not run on hardware
//
#include <hip/hip_runtime.h>
#include <hip/hip_bf16.h>
#include <stdint.h>

// ---------------------------------------------------------------------------
// Problem constants (match the JAX reference)
// ---------------------------------------------------------------------------
#define BB   2
#define CC   64
#define HH   256
#define WW_  256
#define BD   1024            // C*4*4
#define LL   4096            // (256/4)^2
#define HW   65536
#define PH   258             // padded height/width (1-px halo)
#define IMG_ELEMS ((size_t)BB*CC*HH*WW_)   // 8,388,608

typedef __attribute__((ext_vector_type(16))) _Float16 v16h;
typedef __attribute__((ext_vector_type(8)))  _Float16 v8h;
typedef __attribute__((ext_vector_type(8)))  float    v8f;

#define WMMA_F16(a, b, c) \
  __builtin_amdgcn_wmma_f32_16x16x32_f16(false, (a), false, (b), (short)0, (c), false, false)

// ---------------------------------------------------------------------------
// WMMA fragment loaders (CDNA5 ISA §7.12.2, wave32)
// A 16x32 f16: lane(0..15)=row M; elems 0..7 -> k0+8h+e, elems 8..15 -> +16
// B 32x16 f16: lane(0..15)=col N; elems 0..7 -> k0+8h+e, elems 8..15 -> +16
// ---------------------------------------------------------------------------
__device__ __forceinline__ v16h load_fragA(const _Float16* __restrict__ p) {
  v8h lo = *(const v8h*)(p);
  v8h hi = *(const v8h*)(p + 16);
  v16h r;
#pragma unroll
  for (int i = 0; i < 8; ++i) { r[i] = lo[i]; r[i + 8] = hi[i]; }
  return r;
}
__device__ __forceinline__ v16h load_fragB(const _Float16* __restrict__ p) {
  v8h lo = *(const v8h*)(p);
  v8h hi = *(const v8h*)(p + 16);
  v16h r;
#pragma unroll
  for (int i = 0; i < 8; ++i) { r[i] = lo[i]; r[i + 8] = hi[i]; }
  return r;
}

// ---------------------------------------------------------------------------
// Generic f16 GEMM: C[M,N](f32) = scale*(A[M,K] @ Bt[N,K]^T) + bias[n]
// grid=(N/128, M/128), 256 thr = 8 waves (4M x 2N), wave tile 32x64 (2x4 WMMA)
// All fragment pointers are named scalars so addrspace inference keeps every
// load on the global (not flat) path.
// ---------------------------------------------------------------------------
__global__ void __launch_bounds__(256)
gemm_f16_wmma(const _Float16* __restrict__ A, int lda,
              const _Float16* __restrict__ Bt, int ldb,
              float* __restrict__ C, int ldc,
              int K, float scale, const float* __restrict__ bias) {
  const int lane = threadIdx.x & 31;
  const int waveId = threadIdx.x >> 5;
  const int waveN = waveId & 1, waveM = waveId >> 1;
  const int m0 = blockIdx.y * 128 + waveM * 32;
  const int n0 = blockIdx.x * 128 + waveN * 64;
  const int l15 = lane & 15, hsel = lane >> 4;

  const _Float16* pA0 = A + (size_t)(m0 + l15) * lda + hsel * 8;
  const _Float16* pA1 = A + (size_t)(m0 + 16 + l15) * lda + hsel * 8;
  const _Float16* pB0 = Bt + (size_t)(n0 + l15) * ldb + hsel * 8;
  const _Float16* pB1 = pB0 + (size_t)16 * ldb;
  const _Float16* pB2 = pB0 + (size_t)32 * ldb;
  const _Float16* pB3 = pB0 + (size_t)48 * ldb;

  v8f c00 = {}, c01 = {}, c02 = {}, c03 = {};
  v8f c10 = {}, c11 = {}, c12 = {}, c13 = {};

  for (int k = 0; k < K; k += 32) {
    v16h a0 = load_fragA(pA0);
    v16h a1 = load_fragA(pA1);
    __builtin_prefetch(pA0 + 96, 0, 1);
    v16h b0 = load_fragB(pB0);
    v16h b1 = load_fragB(pB1);
    v16h b2 = load_fragB(pB2);
    v16h b3 = load_fragB(pB3);

    c00 = WMMA_F16(a0, b0, c00);  c10 = WMMA_F16(a1, b0, c10);
    c01 = WMMA_F16(a0, b1, c01);  c11 = WMMA_F16(a1, b1, c11);
    c02 = WMMA_F16(a0, b2, c02);  c12 = WMMA_F16(a1, b2, c12);
    c03 = WMMA_F16(a0, b3, c03);  c13 = WMMA_F16(a1, b3, c13);

    pA0 += 32; pA1 += 32; pB0 += 32; pB1 += 32; pB2 += 32; pB3 += 32;
  }

  const int mbase = m0 + hsel * 8;
#pragma unroll
  for (int i = 0; i < 2; ++i) {
    const v8f* row0 = i ? &c10 : &c00;  // c00..c03 / c10..c13 are consecutive
#pragma unroll
    for (int tj = 0; tj < 4; ++tj) {
      v8f cc = (tj == 0) ? (i ? c10 : c00)
             : (tj == 1) ? (i ? c11 : c01)
             : (tj == 2) ? (i ? c12 : c02)
                         : (i ? c13 : c03);
      (void)row0;
      int col = n0 + tj * 16 + l15;
      float bv = bias ? bias[col] : 0.0f;
#pragma unroll
      for (int j = 0; j < 8; ++j) {
        int r = mbase + i * 16 + j;
        C[(size_t)r * ldc + col] = cc[j] * scale + bv;
      }
    }
  }
}

// ---------------------------------------------------------------------------
// conv3x3 (64->64) + bias + ReLU as implicit GEMM over padded NHWC f16 input.
// pad: [B][258][258][64] f16, halo = 0.  out: [B*HW][64] f16 (NHWC).
// wpack: [9][co=64][ci=64] f16.  grid.x = (B*HW)/256; 8 waves x 32 rows;
// wave tile 32M x 64N (full co range).  K = 9 taps * 64 ci.
// ---------------------------------------------------------------------------
__global__ void __launch_bounds__(256)
conv3x3_wmma(const _Float16* __restrict__ pad, const _Float16* __restrict__ wpack,
             const float* __restrict__ bias, _Float16* __restrict__ outp) {
  const int lane = threadIdx.x & 31;
  const int waveId = threadIdx.x >> 5;
  const int m0 = blockIdx.x * 256 + waveId * 32;
  const int l15 = lane & 15, hsel = lane >> 4;

  int mA = m0 + l15;
  int b0i = mA >> 16, p0 = mA & 65535;
  const _Float16* c0p = pad + ((size_t)(b0i * PH + (p0 >> 8)) * PH + (p0 & 255)) * 64 + hsel * 8;
  int mB = m0 + 16 + l15;
  int b1i = mB >> 16, p1 = mB & 65535;
  const _Float16* c1p = pad + ((size_t)(b1i * PH + (p1 >> 8)) * PH + (p1 & 255)) * 64 + hsel * 8;
  const _Float16* wb = wpack + (size_t)l15 * 64 + hsel * 8;

  v8f c00 = {}, c01 = {}, c02 = {}, c03 = {};
  v8f c10 = {}, c11 = {}, c12 = {}, c13 = {};

  for (int t = 0; t < 9; ++t) {
    const int ky = t / 3, kx = t % 3;
    const _Float16* a0b = c0p + (ky * PH + kx) * 64;
    const _Float16* a1b = c1p + (ky * PH + kx) * 64;
    const _Float16* bb = wb + t * 4096;
#pragma unroll
    for (int kk = 0; kk < 64; kk += 32) {
      v16h a0 = load_fragA(a0b + kk);
      v16h a1 = load_fragA(a1b + kk);
      v16h b0 = load_fragB(bb + kk);
      v16h b1 = load_fragB(bb + 16 * 64 + kk);
      v16h b2 = load_fragB(bb + 32 * 64 + kk);
      v16h b3 = load_fragB(bb + 48 * 64 + kk);
      c00 = WMMA_F16(a0, b0, c00);  c10 = WMMA_F16(a1, b0, c10);
      c01 = WMMA_F16(a0, b1, c01);  c11 = WMMA_F16(a1, b1, c11);
      c02 = WMMA_F16(a0, b2, c02);  c12 = WMMA_F16(a1, b2, c12);
      c03 = WMMA_F16(a0, b3, c03);  c13 = WMMA_F16(a1, b3, c13);
    }
  }

  const int mbase = m0 + hsel * 8;
#pragma unroll
  for (int i = 0; i < 2; ++i) {
#pragma unroll
    for (int tj = 0; tj < 4; ++tj) {
      v8f cc = (tj == 0) ? (i ? c10 : c00)
             : (tj == 1) ? (i ? c11 : c01)
             : (tj == 2) ? (i ? c12 : c02)
                         : (i ? c13 : c03);
      int co = tj * 16 + l15;
      float bv = bias[co];
#pragma unroll
      for (int j = 0; j < 8; ++j) {
        int r = mbase + i * 16 + j;
        outp[(size_t)r * 64 + co] = (_Float16)fmaxf(cc[j] + bv, 0.0f);
      }
    }
  }
}

// ---------------------------------------------------------------------------
// Fused conv1x1 + unfold + Linear as one GEMM:
//   proj[b*L + l, n] = sum_{s,ci} T[pixel(b,l,s), ci] * Wc[n, s*64+ci] + bc[n]
// T: NHWC f16 [B*HW][64].  Wc: [1024][1024] f16 (k = s*64+ci).  C: f32.
// grid=(1024/128, 8192/128), 8 waves (4M x 2N), wave tile 32x64.
// ---------------------------------------------------------------------------
__global__ void __launch_bounds__(256)
gemm_unfold_wmma(const _Float16* __restrict__ T, const _Float16* __restrict__ Wc,
                 float* __restrict__ C, const float* __restrict__ bias) {
  const int lane = threadIdx.x & 31;
  const int waveId = threadIdx.x >> 5;
  const int waveN = waveId & 1, waveM = waveId >> 1;
  const int m0 = blockIdx.y * 128 + waveM * 32;
  const int n0 = blockIdx.x * 128 + waveN * 64;
  const int l15 = lane & 15, hsel = lane >> 4;

  int mA = m0 + l15;
  int bA = mA >> 12, lA = mA & 4095;
  const _Float16* t0 = T + ((size_t)bA * HW + ((lA >> 6) * 4) * 256 + (lA & 63) * 4) * 64 + hsel * 8;
  int mB = m0 + 16 + l15;
  int bB = mB >> 12, lB = mB & 4095;
  const _Float16* t1 = T + ((size_t)bB * HW + ((lB >> 6) * 4) * 256 + (lB & 63) * 4) * 64 + hsel * 8;

  const _Float16* pB0 = Wc + (size_t)(n0 + l15) * 1024 + hsel * 8;
  const _Float16* pB1 = pB0 + (size_t)16 * 1024;
  const _Float16* pB2 = pB0 + (size_t)32 * 1024;
  const _Float16* pB3 = pB0 + (size_t)48 * 1024;

  v8f c00 = {}, c01 = {}, c02 = {}, c03 = {};
  v8f c10 = {}, c11 = {}, c12 = {}, c13 = {};

  for (int s = 0; s < 16; ++s) {
    const int po = ((s >> 2) * 256 + (s & 3)) * 64;   // pixel offset within block
#pragma unroll
    for (int kk = 0; kk < 64; kk += 32) {
      v16h a0 = load_fragA(t0 + po + kk);
      v16h a1 = load_fragA(t1 + po + kk);
      v16h b0 = load_fragB(pB0);
      v16h b1 = load_fragB(pB1);
      v16h b2 = load_fragB(pB2);
      v16h b3 = load_fragB(pB3);
      c00 = WMMA_F16(a0, b0, c00);  c10 = WMMA_F16(a1, b0, c10);
      c01 = WMMA_F16(a0, b1, c01);  c11 = WMMA_F16(a1, b1, c11);
      c02 = WMMA_F16(a0, b2, c02);  c12 = WMMA_F16(a1, b2, c12);
      c03 = WMMA_F16(a0, b3, c03);  c13 = WMMA_F16(a1, b3, c13);
      pB0 += 32; pB1 += 32; pB2 += 32; pB3 += 32;
    }
  }

  const int mbase = m0 + hsel * 8;
#pragma unroll
  for (int i = 0; i < 2; ++i) {
#pragma unroll
    for (int tj = 0; tj < 4; ++tj) {
      v8f cc = (tj == 0) ? (i ? c10 : c00)
             : (tj == 1) ? (i ? c11 : c01)
             : (tj == 2) ? (i ? c12 : c02)
                         : (i ? c13 : c03);
      int col = n0 + tj * 16 + l15;
      float bv = bias[col];
#pragma unroll
      for (int j = 0; j < 8; ++j) {
        int r = mbase + i * 16 + j;
        C[(size_t)r * 1024 + col] = cc[j] + bv;
      }
    }
  }
}

// ---------------------------------------------------------------------------
// Small data-movement / pointwise kernels
// ---------------------------------------------------------------------------
__global__ void __launch_bounds__(256)
zero_u32(uint32_t* __restrict__ p, int n) {
  int i = blockIdx.x * 256 + threadIdx.x;
  if (i < n) p[i] = 0u;
}

// NCHW f32 -> padded NHWC f16 interior
__global__ void __launch_bounds__(256)
pad_nhwc_f16(const float* __restrict__ in, _Float16* __restrict__ pad) {
  size_t idx = (size_t)blockIdx.x * 256 + threadIdx.x;   // B*HW*64
  int c = idx & 63;
  int x = (int)(idx >> 6) & 255;
  int yy = (int)(idx >> 14) & 255;
  int b = (int)(idx >> 22);
  float v = in[((size_t)(b * 64 + c)) * HW + yy * 256 + x];
  pad[((size_t)(b * PH + yy + 1) * PH + (x + 1)) * 64 + c] = (_Float16)v;
}

// o_unf f32 [B*L][BD] -> padded NHWC f16 (fold + pad)
__global__ void __launch_bounds__(256)
fold_pad_f16(const float* __restrict__ ou, _Float16* __restrict__ pad) {
  size_t idx = (size_t)blockIdx.x * 256 + threadIdx.x;   // B*HW*64, c fastest
  int c = idx & 63;
  int p = (int)(idx >> 6) & 65535;
  int b = (int)(idx >> 22);
  int yy = p >> 8, x = p & 255;
  int l = (yy >> 2) * 64 + (x >> 2);
  int s = (yy & 3) * 4 + (x & 3);
  float v = ou[((size_t)b * LL + l) * BD + c * 16 + s];
  pad[((size_t)(b * PH + yy + 1) * PH + (x + 1)) * 64 + c] = (_Float16)v;
}

// OIHW f32 conv3x3 weights -> [tap][co][ci] f16
__global__ void __launch_bounds__(256)
repack_w3(const float* __restrict__ w, _Float16* __restrict__ wp) {
  int idx = blockIdx.x * 256 + threadIdx.x;              // 9*64*64 = 36864
  int ci = idx & 63, co = (idx >> 6) & 63, t = idx >> 12;
  wp[t * 4096 + co * 64 + ci] = (_Float16)w[(co * 64 + ci) * 9 + t];
}

// W_comb[n, s*64+ci] = sum_co lw[n, co*16+s] * w2[co, ci]
__global__ void __launch_bounds__(256)
make_comb_w(const float* __restrict__ lw, const float* __restrict__ w2,
            _Float16* __restrict__ Wc) {
  int idx = blockIdx.x * 256 + threadIdx.x;              // 1024*16*64 = 1,048,576
  int ci = idx & 63, s = (idx >> 6) & 15, n = idx >> 10;
  float acc = 0.f;
  for (int co = 0; co < 64; ++co)
    acc = fmaf(lw[(size_t)n * 1024 + co * 16 + s], w2[co * 64 + ci], acc);
  Wc[(size_t)n * 1024 + s * 64 + ci] = (_Float16)acc;
}

// bias_comb[n] = lb[n] + sum_co b2[co] * sum_s lw[n, co*16+s]
__global__ void __launch_bounds__(256)
make_comb_bias(const float* __restrict__ lw, const float* __restrict__ b2,
               const float* __restrict__ lb, float* __restrict__ bc) {
  int n = blockIdx.x * 256 + threadIdx.x;                // 1024
  float acc = lb[n];
  for (int co = 0; co < 64; ++co) {
    float s16 = 0.f;
    for (int s = 0; s < 16; ++s) s16 += lw[(size_t)n * 1024 + co * 16 + s];
    acc = fmaf(b2[co], s16, acc);
  }
  bc[n] = acc;
}

// LayerNorm over last dim (1024), output f16. One block per row.
__global__ void __launch_bounds__(256)
layernorm_f16(const float* __restrict__ in, const float* __restrict__ g,
              const float* __restrict__ beta, _Float16* __restrict__ out) {
  __shared__ float s1[256], s2[256];
  const int row = blockIdx.x;
  const int tid = threadIdx.x;
  const float* rp = in + (size_t)row * BD;
  float a = 0.f, b = 0.f;
#pragma unroll
  for (int i = 0; i < 4; ++i) {
    float v = rp[tid + i * 256];
    a += v; b += v * v;
  }
  s1[tid] = a; s2[tid] = b;
  __syncthreads();
  for (int s = 128; s > 0; s >>= 1) {
    if (tid < s) { s1[tid] += s1[tid + s]; s2[tid] += s2[tid + s]; }
    __syncthreads();
  }
  float mu = s1[0] * (1.0f / BD);
  float var = s2[0] * (1.0f / BD) - mu * mu;
  float rs = rsqrtf(var + 1e-5f);
  _Float16* op = out + (size_t)row * BD;
#pragma unroll
  for (int i = 0; i < 4; ++i) {
    int col = tid + i * 256;
    op[col] = (_Float16)((rp[col] - mu) * rs * g[col] + beta[col]);
  }
}

// Row softmax over 4096 f32 scores -> f16 probabilities. One block per row.
__global__ void __launch_bounds__(256)
softmax_f16(const float* __restrict__ S, _Float16* __restrict__ P) {
  __shared__ float red[256];
  const int row = blockIdx.x;
  const int tid = threadIdx.x;
  const float* rp = S + (size_t)row * LL;
  float mx = -3.4e38f;
#pragma unroll
  for (int i = 0; i < 16; ++i) mx = fmaxf(mx, rp[tid + i * 256]);
  red[tid] = mx; __syncthreads();
  for (int s = 128; s > 0; s >>= 1) {
    if (tid < s) red[tid] = fmaxf(red[tid], red[tid + s]);
    __syncthreads();
  }
  mx = red[0]; __syncthreads();
  float sum = 0.f;
  float e[16];
#pragma unroll
  for (int i = 0; i < 16; ++i) { e[i] = __expf(rp[tid + i * 256] - mx); sum += e[i]; }
  red[tid] = sum; __syncthreads();
  for (int s = 128; s > 0; s >>= 1) {
    if (tid < s) red[tid] += red[tid + s];
    __syncthreads();
  }
  float inv = 1.0f / red[0];
  _Float16* op = P + (size_t)row * LL;
#pragma unroll
  for (int i = 0; i < 16; ++i) op[tid + i * 256] = (_Float16)(e[i] * inv);
}

// Transpose V: f32 [4096,1024] -> f16 [1024,4096] (z = batch)
__global__ void __launch_bounds__(1024)
transpose_f32_to_f16(const float* __restrict__ in, _Float16* __restrict__ out) {
  __shared__ float tile[32][33];
  const int b = blockIdx.z;
  const float* ip = in + (size_t)b * LL * BD;
  _Float16* op = out + (size_t)b * BD * LL;
  int m = blockIdx.y * 32 + threadIdx.y;
  int n = blockIdx.x * 32 + threadIdx.x;
  tile[threadIdx.y][threadIdx.x] = ip[(size_t)m * BD + n];
  __syncthreads();
  int on = blockIdx.x * 32 + threadIdx.y;
  int om = blockIdx.y * 32 + threadIdx.x;
  op[(size_t)on * LL + om] = (_Float16)tile[threadIdx.x][threadIdx.y];
}

// final conv1x1 + residual, reading NHWC f16, writing NCHW f32
__global__ void __launch_bounds__(256)
conv1x1_residual_nhwc(const _Float16* __restrict__ t, const float* __restrict__ w2,
                      const float* __restrict__ b2, const float* __restrict__ xres,
                      float* __restrict__ out) {
  size_t idx = (size_t)blockIdx.x * 256 + threadIdx.x;   // NCHW order
  int x = idx & 255;
  int yy = (idx >> 8) & 255;
  int c = (idx >> 16) & 63;
  int b = (int)(idx >> 22);
  const _Float16* tp = t + ((size_t)b * HW + yy * 256 + x) * 64;
  const float* wp = w2 + c * 64;
  float acc = b2[c];
  for (int ci = 0; ci < 64; ++ci)
    acc = fmaf(wp[ci], (float)tp[ci], acc);
  out[idx] = xres[idx] + 1.0f * acc;                     // ALPHA = 1.0
}

// ---------------------------------------------------------------------------
// Host side
// ---------------------------------------------------------------------------
static inline void launch_gemm(const _Float16* A, int lda, const _Float16* Bt, int ldb,
                               float* C, int ldc, int M, int N, int K,
                               float scale, const float* bias, hipStream_t s) {
  dim3 g(N / 128, M / 128);
  gemm_f16_wmma<<<g, 256, 0, s>>>(A, lda, Bt, ldb, C, ldc, K, scale, bias);
}

extern "C" void kernel_launch(void* const* d_in, const int* in_sizes, int n_in,
                              void* d_out, int out_size, void* d_ws, size_t ws_size,
                              hipStream_t stream) {
  (void)in_sizes; (void)n_in; (void)out_size; (void)ws_size;

  const float* x    = (const float*)d_in[0];
  const float* y    = (const float*)d_in[1];
  const float* ln_g = (const float*)d_in[20];
  const float* ln_b = (const float*)d_in[21];
  const float* wo1  = (const float*)d_in[22];
  const float* bo1  = (const float*)d_in[23];
  const float* wo2  = (const float*)d_in[24];
  const float* bo2  = (const float*)d_in[25];
  float* out = (float*)d_out;

  // ---- workspace layout (all offsets multiples of 256B) ----
  const size_t SZ_PAD_F16 = (size_t)BB * PH * PH * 64 * 2;  // 17,040,384
  const size_t SZ_NHWC    = (size_t)BB * HW * 64 * 2;       // 16,777,216
  const size_t SZ_PROJ    = (size_t)BB * LL * BD * 4;       // 33,554,432
  const size_t SZ_UNF_F16 = (size_t)BB * LL * BD * 2;       // 16,777,216
  const size_t SZ_S_F32   = (size_t)LL * LL * 4;            // 67,108,864
  const size_t SZ_P_F16   = (size_t)LL * LL * 2;            // 33,554,432
  char* ws = (char*)d_ws;
  size_t off = 0;
  _Float16* x_pad  = (_Float16*)(ws + off); off += SZ_PAD_F16;
  _Float16* y_pad  = (_Float16*)(ws + off); off += SZ_PAD_F16;
  _Float16* o_pad  = (_Float16*)(ws + off); off += SZ_PAD_F16;
  _Float16* wpack  = (_Float16*)(ws + off); off += (size_t)9 * 4096 * 2;     // 73,728
  _Float16* t_nhwc = (_Float16*)(ws + off); off += SZ_NHWC;
  _Float16* t_o    = (_Float16*)(ws + off); off += SZ_NHWC;
  _Float16* Wc     = (_Float16*)(ws + off); off += (size_t)BD * BD * 2;      // 2 MB
  float*    bc     = (float*)(ws + off);    off += 4096;
  float*    proj   = (float*)(ws + off);    off += SZ_PROJ;
  _Float16* qn     = (_Float16*)(ws + off); off += SZ_UNF_F16;
  _Float16* kn     = (_Float16*)(ws + off); off += SZ_UNF_F16;
  _Float16* vT     = (_Float16*)(ws + off); off += SZ_UNF_F16;
  float*    scores = (float*)(ws + off);    off += SZ_S_F32;
  _Float16* P      = (_Float16*)(ws + off); off += SZ_P_F16;
  float*    o_unf  = (float*)(ws + off);    off += SZ_PROJ;

  const int IMG_BLOCKS = (int)(IMG_ELEMS / 256);            // 32768
  const int PAD_U32    = (int)(SZ_PAD_F16 / 4);             // 4,260,096
  const int PAD_ZB     = (PAD_U32 + 255) / 256;

  // ---- build padded NHWC f16 inputs (halo zeroed every call) ----
  zero_u32<<<PAD_ZB, 256, 0, stream>>>((uint32_t*)x_pad, PAD_U32);
  pad_nhwc_f16<<<IMG_BLOCKS, 256, 0, stream>>>(x, x_pad);
  zero_u32<<<PAD_ZB, 256, 0, stream>>>((uint32_t*)y_pad, PAD_U32);
  pad_nhwc_f16<<<IMG_BLOCKS, 256, 0, stream>>>(y, y_pad);

  // ---- q / k / v projection pipelines (all WMMA) ----
  struct Proj { const _Float16* pad; const float *w1, *b1, *w2, *b2, *lw, *lb; };
  Proj pj[3] = {
    { x_pad, (const float*)d_in[2],  (const float*)d_in[3],  (const float*)d_in[4],
             (const float*)d_in[5],  (const float*)d_in[6],  (const float*)d_in[7]  },
    { y_pad, (const float*)d_in[8],  (const float*)d_in[9],  (const float*)d_in[10],
             (const float*)d_in[11], (const float*)d_in[12], (const float*)d_in[13] },
    { y_pad, (const float*)d_in[14], (const float*)d_in[15], (const float*)d_in[16],
             (const float*)d_in[17], (const float*)d_in[18], (const float*)d_in[19] },
  };

  for (int p = 0; p < 3; ++p) {
    repack_w3<<<144, 256, 0, stream>>>(pj[p].w1, wpack);
    conv3x3_wmma<<<(BB * HW) / 256, 256, 0, stream>>>(pj[p].pad, wpack, pj[p].b1, t_nhwc);
    make_comb_w<<<4096, 256, 0, stream>>>(pj[p].lw, pj[p].w2, Wc);
    make_comb_bias<<<4, 256, 0, stream>>>(pj[p].lw, pj[p].b2, pj[p].lb, bc);
    gemm_unfold_wmma<<<dim3(BD / 128, (BB * LL) / 128), 256, 0, stream>>>(t_nhwc, Wc, proj, bc);
    if (p == 0) {
      layernorm_f16<<<BB * LL, 256, 0, stream>>>(proj, ln_g, ln_b, qn);
    } else if (p == 1) {
      layernorm_f16<<<BB * LL, 256, 0, stream>>>(proj, ln_g, ln_b, kn);
    } else {
      dim3 tg(BD / 32, LL / 32, BB);
      transpose_f32_to_f16<<<tg, dim3(32, 32), 0, stream>>>(proj, vT);
    }
  }

  // ---- attention per batch (WMMA score GEMM -> softmax -> WMMA PV GEMM) ----
  const float inv_sqrt_bd = 0.03125f;  // 1024^-0.5
  for (int b = 0; b < BB; ++b) {
    const _Float16* qb = qn + (size_t)b * LL * BD;
    const _Float16* kb = kn + (size_t)b * LL * BD;
    const _Float16* vb = vT + (size_t)b * BD * LL;
    float* ob = o_unf + (size_t)b * LL * BD;
    launch_gemm(qb, BD, kb, BD, scores, LL, LL, LL, BD, inv_sqrt_bd, nullptr, stream);
    softmax_f16<<<LL, 256, 0, stream>>>(scores, P);
    launch_gemm(P, LL, vb, LL, ob, BD, LL, BD, LL, 1.0f, nullptr, stream);
  }

  // ---- fold -> padded NHWC, output conv3x3 (WMMA) + conv1x1 + residual ----
  zero_u32<<<PAD_ZB, 256, 0, stream>>>((uint32_t*)o_pad, PAD_U32);
  fold_pad_f16<<<IMG_BLOCKS, 256, 0, stream>>>(o_unf, o_pad);
  repack_w3<<<144, 256, 0, stream>>>(wo1, wpack);
  conv3x3_wmma<<<(BB * HW) / 256, 256, 0, stream>>>(o_pad, wpack, bo1, t_o);
  conv1x1_residual_nhwc<<<IMG_BLOCKS, 256, 0, stream>>>(t_o, wo2, bo2, x, out);
}
